// CrossAttention_8735963480768
// MI455X (gfx1250) — matrix operand order repaired and checked by reading, not compiled, not run on hardware
//
#include <hip/hip_runtime.h>
#include <stdint.h>

typedef __attribute__((ext_vector_type(16))) __bf16 v16bf;
typedef __attribute__((ext_vector_type(8)))  float  v8f;

#define NB   8
#define TQ   100
#define TS   100
#define EE   512
#define DD   512
#define K2E  1024          // 2E
#define ASTR 1032          // padded LDS row stride (bf16 elems): 2064B -> conflict-free b128 frag loads
#define MPAD 112           // 7 m-tiles of 16 (covers 100 rows)

union FragAB { uint4 q[2]; v16bf v; };

__device__ __forceinline__ unsigned short f2bf(float f) {
  unsigned int u = __float_as_uint(f);
  u += 0x7FFFu + ((u >> 16) & 1u);     // round-to-nearest-even
  return (unsigned short)(u >> 16);
}

// ---------------------------------------------------------------------------
// Pre-pass: W1 [1024,512] f32 row-major -> bf16 tiles pre-swizzled into the
// exact WMMA B-fragment lane layout (lane's 16 values contiguous -> 2x b128).
// Tile index = nt*32 + kt; 1 KB per tile; 1 MB total (stays resident in L2).
// ---------------------------------------------------------------------------
__global__ void w1_swizzle_kernel(const float* __restrict__ W1,
                                  unsigned short* __restrict__ wsB) {
  const int tile = blockIdx.x;             // 0..1023
  const int nt = tile >> 5, kt = tile & 31;
  const int l = threadIdx.x;               // 0..31 (one wave)
  const int col   = nt * 16 + (l & 15);
  const int kbase = kt * 32 + ((l >> 4) << 4);   // lanes 0-15: K 0..15, lanes 16-31: K 16..31
  unsigned short* o = wsB + (size_t)tile * 512 + l * 16;
#pragma unroll
  for (int j = 0; j < 16; ++j)
    o[j] = f2bf(W1[(size_t)(kbase + j) * DD + col]);
}

// ---------------------------------------------------------------------------
// Per-tile epilogue: tanh(acc + b1)*w2, row-sum across 16 lanes, accumulate
// partial scores into LDS.
// ---------------------------------------------------------------------------
__device__ __forceinline__ void tile_reduce(v8f acc, float bias, float wt,
                                            int mt, int lane, float* sScores) {
  union { v8f v; float f[8]; } u; u.v = acc;
  const int rbase = mt * 16 + ((lane >> 4) << 3);
#pragma unroll
  for (int v = 0; v < 8; ++v) {
    float x = tanhf(u.f[v] + bias) * wt;
    x += __shfl_xor(x, 1, 32);
    x += __shfl_xor(x, 2, 32);
    x += __shfl_xor(x, 4, 32);
    x += __shfl_xor(x, 8, 32);
    if ((lane & 15) == 0) atomicAdd(&sScores[rbase + v], x);
  }
}

// ---------------------------------------------------------------------------
// Main kernel: one workgroup (8 wave32) per (b,q).
// ---------------------------------------------------------------------------
__global__ void __launch_bounds__(256)
xattn_kernel(const float* __restrict__ src, const float* __restrict__ tgt,
             const float* __restrict__ mask, const float* __restrict__ b1,
             const float* __restrict__ w2, const unsigned short* __restrict__ wsB,
             float* __restrict__ out) {
  extern __shared__ unsigned char smem[];
  unsigned short* sA = (unsigned short*)smem;                       // MPAD*ASTR bf16
  float* sB1     = (float*)(smem + (size_t)MPAD * ASTR * 2);        // 512
  float* sW2     = sB1 + 512;                                       // 512
  float* sScores = sW2 + 512;                                       // 112
  float* sW      = sScores + 112;                                   // 112
  float* sRed    = sW + 112;                                        // 256

  const int tid = threadIdx.x;
  const int bq  = blockIdx.x;                                       // 0..799

  // ---- stage b1/w2, zero scores ----
  sB1[tid] = b1[tid]; sB1[tid + 256] = b1[tid + 256];
  sW2[tid] = w2[tid]; sW2[tid + 256] = w2[tid + 256];
  if (tid < MPAD) sScores[tid] = 0.f;

  // ---- stage A = cat(src,tgt)[b,q] as bf16 into LDS (row-major, padded) ----
  const float4* s4 = (const float4*)(src + (size_t)bq * TS * EE);
  const float4* t4 = (const float4*)(tgt + (size_t)bq * TS * EE);
  for (int i = tid; i < TS * (EE / 4); i += 256) {                  // 12800 float4s
    const int row = i >> 7, c4 = i & 127;
    float4 a = s4[row * 128 + c4];
    unsigned int lo = f2bf(a.x) | ((unsigned int)f2bf(a.y) << 16);
    unsigned int hi = f2bf(a.z) | ((unsigned int)f2bf(a.w) << 16);
    *(uint2*)(sA + row * ASTR + c4 * 4) = make_uint2(lo, hi);
    float4 b = t4[row * 128 + c4];
    lo = f2bf(b.x) | ((unsigned int)f2bf(b.y) << 16);
    hi = f2bf(b.z) | ((unsigned int)f2bf(b.w) << 16);
    *(uint2*)(sA + row * ASTR + 512 + c4 * 4) = make_uint2(lo, hi);
  }
  for (int i = tid; i < 12 * 1024; i += 256) {                      // zero pad rows 100..111
    const int row = 100 + (i >> 10), c = i & 1023;
    sA[row * ASTR + c] = 0;
  }
  __syncthreads();

  // ---- GEMM: h = A[112x1024] @ W1[1024x512] via v_wmma_f32_16x16x32_bf16 ----
  const int wave = tid >> 5, lane = tid & 31;
  const int laneRow = lane & 15;
  const int laneK   = (lane >> 4) << 4;
  const unsigned short* aBase = sA + laneRow * ASTR + laneK;

#pragma unroll 1
  for (int pass = 0; pass < 2; ++pass) {
    const int nt0 = wave * 4 + pass * 2;
    const int nt1 = nt0 + 1;
    v8f acc0[7], acc1[7];
#pragma unroll
    for (int mt = 0; mt < 7; ++mt) {
      acc0[mt] = (v8f){0.f,0.f,0.f,0.f,0.f,0.f,0.f,0.f};
      acc1[mt] = (v8f){0.f,0.f,0.f,0.f,0.f,0.f,0.f,0.f};
    }
    const uint4* bp0 = (const uint4*)(wsB + (size_t)(nt0 * 32) * 512) + lane * 2;
    const uint4* bp1 = (const uint4*)(wsB + (size_t)(nt1 * 32) * 512) + lane * 2;
#pragma unroll 1
    for (int kt = 0; kt < 32; ++kt) {
      FragAB fb0, fb1;
      fb0.q[0] = bp0[0]; fb0.q[1] = bp0[1]; bp0 += 32;   // advance 32 x b128 = 256 bf16 (half a K-tile)
      fb1.q[0] = bp1[0]; fb1.q[1] = bp1[1]; bp1 += 32;
      const unsigned short* ak = aBase + kt * 32;
#pragma unroll
      for (int mt = 0; mt < 7; ++mt) {
        FragAB fa;
        const uint4* ap = (const uint4*)(ak + mt * 16 * ASTR);
        fa.q[0] = ap[0];           // K = kt*32 + laneK + [0..7]
        fa.q[1] = ap[1];           // K = kt*32 + laneK + [8..15]
        acc0[mt] = __builtin_amdgcn_wmma_f32_16x16x32_bf16(
            false, fa.v, false, fb0.v, (short)0, acc0[mt], false, false);
        acc1[mt] = __builtin_amdgcn_wmma_f32_16x16x32_bf16(
            false, fa.v, false, fb1.v, (short)0, acc1[mt], false, false);
      }
    }
    // ---- epilogue: scores += sum_n tanh(h+b1)*w2 ----
    const float bias0 = sB1[nt0 * 16 + laneRow], wt0 = sW2[nt0 * 16 + laneRow];
    const float bias1 = sB1[nt1 * 16 + laneRow], wt1 = sW2[nt1 * 16 + laneRow];
#pragma unroll
    for (int mt = 0; mt < 7; ++mt) {
      tile_reduce(acc0[mt], bias0, wt0, mt, lane, sScores);
      tile_reduce(acc1[mt], bias1, wt1, mt, lane, sScores);
    }
  }
  __syncthreads();

  // ---- masked softmax over TS=100 ----
  float sc = 0.f;
  if (tid < TS) sc = sScores[tid] * mask[(size_t)bq * TS + tid];
  sRed[tid] = (tid < TS) ? sc : -3.4e38f;
  __syncthreads();
  for (int off = 128; off > 0; off >>= 1) {
    if (tid < off) sRed[tid] = fmaxf(sRed[tid], sRed[tid + off]);
    __syncthreads();
  }
  const float m = sRed[0];
  const float e = (tid < TS) ? __expf(sc - m) : 0.f;
  __syncthreads();
  sRed[tid] = e;
  __syncthreads();
  for (int off = 128; off > 0; off >>= 1) {
    if (tid < off) sRed[tid] += sRed[tid + off];
    __syncthreads();
  }
  const float S = sRed[0];
  if (tid < TS) {
    const float w = e / S;
    sW[tid] = w;
    out[(size_t)(NB * TQ * EE) + (size_t)bq * TS + tid] = w;        // attn_weights
  }
  __syncthreads();

  // ---- attn_out[b,q,:] = sum_s w[s] * src[b,q,s,:]  (exact fp32, coalesced) ----
  const float2* s2 = (const float2*)(src + (size_t)bq * TS * EE);
  float2 acc = make_float2(0.f, 0.f);
  for (int s = 0; s < TS; ++s) {
    const float w = sW[s];
    const float2 v = s2[s * 256 + tid];
    acc.x += w * v.x; acc.y += w * v.y;
  }
  ((float2*)(out + (size_t)bq * EE))[tid] = acc;
}

// ---------------------------------------------------------------------------
extern "C" void kernel_launch(void* const* d_in, const int* in_sizes, int n_in,
                              void* d_out, int out_size, void* d_ws, size_t ws_size,
                              hipStream_t stream) {
  const float* src  = (const float*)d_in[0];
  const float* tgt  = (const float*)d_in[1];
  const float* mask = (const float*)d_in[2];
  const float* W1   = (const float*)d_in[3];
  const float* b1   = (const float*)d_in[4];
  const float* w2   = (const float*)d_in[5];
  unsigned short* wsB = (unsigned short*)d_ws;    // 1 MB: swizzled bf16 W1

  w1_swizzle_kernel<<<1024, 32, 0, stream>>>(W1, wsB);

  const size_t smemBytes = (size_t)MPAD * ASTR * 2   // A (bf16)
                         + 512 * 4 + 512 * 4          // b1, w2
                         + 112 * 4 + 112 * 4          // scores, weights
                         + 256 * 4;                   // reduction buffer
  xattn_kernel<<<NB * TQ, 256, smemBytes, stream>>>(src, tgt, mask, b1, w2, wsB,
                                                    (float*)d_out);
}
